// PairMultiHeadAttention_86328842649686
// MI455X (gfx1250) — matrix operand order repaired and checked by reading, not compiled, not run on hardware
//
#include <hip/hip_runtime.h>
#include <hip/hip_bf16.h>

typedef __attribute__((ext_vector_type(16))) _Float16 v16h;
typedef __attribute__((ext_vector_type(8)))  float    v8f;

#define TM 128
#define TN 128
#define TK 32
#define PADA 4   // A row stride 36 halves (18 dwords) -> conflict-free m-gather
#define PADB 4   // B row stride 132 halves (66 dwords) -> conflict-free k-gather

static __device__ __forceinline__ uint2 pack4h(float a, float b, float c, float d)
{
    union { _Float16 h[4]; uint2 u; } p;
    p.h[0] = (_Float16)a; p.h[1] = (_Float16)b;
    p.h[2] = (_Float16)c; p.h[3] = (_Float16)d;
    return p.u;
}

// ---------------------------------------------------------------------------
// Generic batched WMMA GEMM:  C[z] = scale * (A[z] @ op(W[z])) + bias + add
//   op(W) = W (K x N, row stride ldw)              when transB == 0
//   op(W) = W^T where W is (N x K, row stride ldw) when transB == 1
// Batch z = i1*n2 + i2 with independent strides per level.
// 128x128 block tile, 16 waves, each wave 32x32 (2x2 WMMA accumulators).
// 2-deep software pipeline: global loads for tile k+1 are issued into VGPRs,
// WMMAs for tile k execute from ping-pong LDS while those loads are in flight,
// then the staged registers are converted to f16 and stored to the other LDS
// buffer (DS ops are in-order per wave, so stores must come AFTER compute).
// ---------------------------------------------------------------------------
__global__ __launch_bounds__(512)
void wmma_gemm(const float* __restrict__ A, const float* __restrict__ W,
               const float* __restrict__ bias, const float* __restrict__ add,
               float* __restrict__ C,
               int M, int N, int K,
               int lda, int ldw, int ldc,
               int transB, float scale, int n2,
               long long sA1, long long sA2,
               long long sW1, long long sW2,
               long long sC1, long long sC2)
{
    int z  = blockIdx.z;
    int i1 = z / n2;
    int i2 = z - i1 * n2;
    A += i1 * sA1 + i2 * sA2;
    W += i1 * sW1 + i2 * sW2;
    long long cOff = i1 * sC1 + i2 * sC2;
    C += cOff;
    if (add) add += cOff;

    __shared__ _Float16 sAz[2][TM][TK + PADA];
    __shared__ _Float16 sBz[2][TK][TN + PADB];

    const int tid  = threadIdx.x;
    const int mb   = blockIdx.y * TM;
    const int nb   = blockIdx.x * TN;
    const int wid  = tid >> 5;
    const int lane = tid & 31;
    const int wy   = wid >> 2;   // 0..3 : 32-row band
    const int wx   = wid & 3;    // 0..3 : 32-col band
    const int lm   = lane & 15;
    const int lhi  = lane >> 4;

    v8f acc[2][2];
    #pragma unroll
    for (int mi = 0; mi < 2; ++mi)
        #pragma unroll
        for (int ni = 0; ni < 2; ++ni)
            acc[mi][ni] = (v8f){0.f,0.f,0.f,0.f,0.f,0.f,0.f,0.f};

    float ra[8], rb[8];   // per-thread staging registers (one tile each)

    // ---- issue global loads of one A/W tile into registers (no waiting) ----
    auto loadA = [&](int k0) {
        if ((k0 + TK <= K) && (mb + TM <= M)) {
            #pragma unroll
            for (int it = 0; it < 2; ++it) {          // 1024 float4 / 512 thr
                int idx = tid + it * 512;
                int r = idx >> 3, c4 = idx & 7;       // 8 float4 per row
                const float4 v = *(const float4*)(A + (long long)(mb + r) * lda + k0 + c4 * 4);
                ra[it*4+0] = v.x; ra[it*4+1] = v.y; ra[it*4+2] = v.z; ra[it*4+3] = v.w;
            }
        } else {
            #pragma unroll
            for (int it = 0; it < 8; ++it) {          // 4096 elems / 512 thr
                int idx = tid + it * 512;
                int r = idx >> 5, c = idx & 31;
                int gm = mb + r, gk = k0 + c;
                bool ok = (gm < M) && (gk < K);
                ra[it] = A[ok ? ((long long)gm * lda + gk) : 0];  // clamped addr
            }
        }
    };
    auto loadB = [&](int k0) {
        const bool bFast = (k0 + TK <= K) && (nb + TN <= N);
        if (bFast && !transB) {
            #pragma unroll
            for (int it = 0; it < 2; ++it) {          // 1024 float4 / 512 thr
                int idx = tid + it * 512;
                int r = idx >> 5, c4 = idx & 31;      // 32 float4 per row
                const float4 v = *(const float4*)(W + (long long)(k0 + r) * ldw + nb + c4 * 4);
                rb[it*4+0] = v.x; rb[it*4+1] = v.y; rb[it*4+2] = v.z; rb[it*4+3] = v.w;
            }
        } else if (bFast) {                           // transB: vector along K
            #pragma unroll
            for (int it = 0; it < 2; ++it) {
                int idx = tid + it * 512;
                int n = idx >> 3, k4 = idx & 7;       // 8 float4 per n-row
                const float4 v = *(const float4*)(W + (long long)(nb + n) * ldw + k0 + k4 * 4);
                rb[it*4+0] = v.x; rb[it*4+1] = v.y; rb[it*4+2] = v.z; rb[it*4+3] = v.w;
            }
        } else {
            #pragma unroll
            for (int it = 0; it < 8; ++it) {
                int idx = tid + it * 512;
                int r = idx >> 7, c = idx & 127;
                int gk = k0 + r, gn = nb + c;
                bool ok = (gk < K) && (gn < N);
                long long o = ok ? (transB ? (long long)gn * ldw + gk
                                           : (long long)gk * ldw + gn)
                                 : 0;
                rb[it] = W[o];
            }
        }
    };

    // ---- convert staged registers to f16 and store into LDS buffer p ----
    auto storeA = [&](int k0, int p) {
        if ((k0 + TK <= K) && (mb + TM <= M)) {
            #pragma unroll
            for (int it = 0; it < 2; ++it) {
                int idx = tid + it * 512;
                int r = idx >> 3, c4 = idx & 7;
                *(uint2*)&sAz[p][r][c4 * 4] =
                    pack4h(ra[it*4+0], ra[it*4+1], ra[it*4+2], ra[it*4+3]);
            }
        } else {
            #pragma unroll
            for (int it = 0; it < 8; ++it) {
                int idx = tid + it * 512;
                int r = idx >> 5, c = idx & 31;
                int gm = mb + r, gk = k0 + c;
                bool ok = (gm < M) && (gk < K);
                sAz[p][r][c] = (_Float16)(ok ? ra[it] : 0.f);
            }
        }
    };
    auto storeB = [&](int k0, int p) {
        const bool bFast = (k0 + TK <= K) && (nb + TN <= N);
        if (bFast && !transB) {
            #pragma unroll
            for (int it = 0; it < 2; ++it) {
                int idx = tid + it * 512;
                int r = idx >> 5, c4 = idx & 31;
                *(uint2*)&sBz[p][r][c4 * 4] =
                    pack4h(rb[it*4+0], rb[it*4+1], rb[it*4+2], rb[it*4+3]);
            }
        } else if (bFast) {
            #pragma unroll
            for (int it = 0; it < 2; ++it) {
                int idx = tid + it * 512;
                int n = idx >> 3, k4 = idx & 7;
                sBz[p][k4 * 4 + 0][n] = (_Float16)rb[it*4+0];
                sBz[p][k4 * 4 + 1][n] = (_Float16)rb[it*4+1];
                sBz[p][k4 * 4 + 2][n] = (_Float16)rb[it*4+2];
                sBz[p][k4 * 4 + 3][n] = (_Float16)rb[it*4+3];
            }
        } else {
            #pragma unroll
            for (int it = 0; it < 8; ++it) {
                int idx = tid + it * 512;
                int r = idx >> 7, c = idx & 127;
                int gk = k0 + r, gn = nb + c;
                bool ok = (gk < K) && (gn < N);
                sBz[p][r][c] = (_Float16)(ok ? rb[it] : 0.f);
            }
        }
    };

    const int nK = (K + TK - 1) / TK;

    // prologue: stage tile 0 into buffer 0
    loadA(0); loadB(0);
    storeA(0, 0); storeB(0, 0);
    __syncthreads();

    for (int kt = 0; kt < nK; ++kt) {
        const int  p    = kt & 1;
        const int  k1   = (kt + 1) * TK;
        const bool more = (kt + 1) < nK;

        // issue next tile's global loads (results not consumed until after compute)
        if (more) { loadA(k1); loadB(k1); }

        // prefetch tile kt+2 into cache (global_prefetch_b8)
        if (kt + 2 < nK) {
            int k2 = (kt + 2) * TK;
            __builtin_prefetch(A + (long long)(mb + (tid & 127)) * lda + k2, 0, 3);
            if (!transB)
                __builtin_prefetch(W + (long long)(k2 + (tid & 31)) * ldw + nb, 0, 3);
            else
                __builtin_prefetch(W + (long long)(nb + (tid & 127)) * ldw + k2, 0, 3);
        }

        // ---- compute tile kt from LDS buffer p: 2x2 WMMA per wave ----
        {
            v16h af0, af1, bf0, bf1;
            const int m0 = wy * 32 + lm;
            const int m1 = m0 + 16;
            #pragma unroll
            for (int e = 0; e < 16; ++e) {
                int r  = e >> 1;
                int kk = ((r < 4) ? (2 * r) : (16 + 2 * (r - 4))) + lhi * 8 + (e & 1);
                af0[e] = sAz[p][m0][kk];
                af1[e] = sAz[p][m1][kk];
            }
            const int nc0 = wx * 32;
            #pragma unroll
            for (int e = 0; e < 16; ++e) {
                int kb = ((e < 8) ? e : (e + 8)) + lhi * 8;
                bf0[e] = sBz[p][kb][nc0 + lm];
                bf1[e] = sBz[p][kb][nc0 + 16 + lm];
            }
            acc[0][0] = __builtin_amdgcn_wmma_f32_16x16x32_f16(false, af0, false, bf0, (short)0, acc[0][0], false, false);
            acc[0][1] = __builtin_amdgcn_wmma_f32_16x16x32_f16(false, af0, false, bf1, (short)0, acc[0][1], false, false);
            acc[1][0] = __builtin_amdgcn_wmma_f32_16x16x32_f16(false, af1, false, bf0, (short)0, acc[1][0], false, false);
            acc[1][1] = __builtin_amdgcn_wmma_f32_16x16x32_f16(false, af1, false, bf1, (short)0, acc[1][1], false, false);
        }

        // now commit the next tile into the other buffer (waits on its loads)
        if (more) { storeA(k1, p ^ 1); storeB(k1, p ^ 1); }
        __syncthreads();
    }

    // C/D layout: lanes 0-15 -> N=lane, M=r; lanes 16-31 -> N=lane-16, M=r+8
    #pragma unroll
    for (int mi = 0; mi < 2; ++mi) {
        #pragma unroll
        for (int ni = 0; ni < 2; ++ni) {
            #pragma unroll
            for (int r = 0; r < 8; ++r) {
                int gm = mb + wy * 32 + mi * 16 + lhi * 8 + r;
                int gn = nb + wx * 32 + ni * 16 + lm;
                if (gm < M && gn < N) {
                    float v = acc[mi][ni][r] * scale;
                    if (bias) v += bias[gn];
                    long long ci = (long long)gm * ldc + gn;
                    if (add) v += add[ci];
                    C[ci] = v;
                }
            }
        }
    }
}

// ---------------------------------------------------------------------------
// Row softmax with optional causal mask. One block per row, L <= 128.
// ---------------------------------------------------------------------------
__global__ __launch_bounds__(128)
void softmax_rows(float* __restrict__ x, int L, int Sdim, int causal)
{
    long long row = blockIdx.x;
    float* p = x + row * L;
    int q = (int)(row % Sdim);
    int valid = causal ? (q + 1) : L;
    int j = threadIdx.x;

    __shared__ float red[128];
    float v = (j < valid) ? p[j] : -3.4e38f;
    red[j] = v;
    __syncthreads();
    #pragma unroll
    for (int s = 64; s > 0; s >>= 1) {
        if (j < s) red[j] = fmaxf(red[j], red[j + s]);
        __syncthreads();
    }
    float mx = red[0];
    __syncthreads();
    float e = (j < valid) ? __expf(v - mx) : 0.0f;
    red[j] = e;
    __syncthreads();
    #pragma unroll
    for (int s = 64; s > 0; s >>= 1) {
        if (j < s) red[j] += red[j + s];
        __syncthreads();
    }
    float inv = 1.0f / red[0];
    if (j < L) p[j] = e * inv;
}

// ---------------------------------------------------------------------------
// Row LayerNorm: y = (x - mean) * rsqrt(var + eps) * g + b. One block/row.
// ---------------------------------------------------------------------------
__global__ __launch_bounds__(256)
void ln_rows(const float* __restrict__ x, const float* __restrict__ g,
             const float* __restrict__ b, float* __restrict__ y,
             int N, float eps)
{
    long long row = blockIdx.x;
    const float* p = x + row * N;
    float* o = y + row * N;
    int t = threadIdx.x;
    float s = 0.f, s2 = 0.f;
    for (int j = t; j < N; j += 256) { float v = p[j]; s += v; s2 += v * v; }
    __shared__ float rs[256], rq[256];
    rs[t] = s; rq[t] = s2;
    __syncthreads();
    #pragma unroll
    for (int k = 128; k > 0; k >>= 1) {
        if (t < k) { rs[t] += rs[t + k]; rq[t] += rq[t + k]; }
        __syncthreads();
    }
    float mean = rs[0] / N;
    float var  = rq[0] / N - mean * mean;
    float inv  = rsqrtf(var + eps);
    for (int j = t; j < N; j += 256)
        o[j] = (p[j] - mean) * inv * g[j] + b[j];
}

// ---------------------------------------------------------------------------
// MOE combine: out[bs,d] = sum_e eoW[e][bs,d] * gate[bs,e]   (NE == 3)
// ---------------------------------------------------------------------------
__global__ __launch_bounds__(256)
void moe_combine(const float* __restrict__ eoW, const float* __restrict__ gate,
                 float* __restrict__ out, int total, long long perE)
{
    int i = blockIdx.x * 256 + threadIdx.x;
    if (i >= total) return;
    int bs = i >> 6;  // DS = 64
    float g0 = gate[bs * 3 + 0], g1 = gate[bs * 3 + 1], g2 = gate[bs * 3 + 2];
    out[i] = eoW[i] * g0 + eoW[perE + i] * g1 + eoW[2 * perE + i] * g2;
}

// ---------------------------------------------------------------------------
extern "C" void kernel_launch(void* const* d_in, const int* in_sizes, int n_in,
                              void* d_out, int out_size, void* d_ws, size_t ws_size,
                              hipStream_t stream)
{
    (void)in_sizes; (void)n_in; (void)out_size; (void)ws_size;

    const float* hidden = (const float*)d_in[0];
    const float* tag    = (const float*)d_in[1];
    const float* feats  = (const float*)d_in[2];
    const float* refs   = (const float*)d_in[3];
    const float* W_moe  = (const float*)d_in[4];
    const float* b_moe  = (const float*)d_in[5];
    const float* Wg     = (const float*)d_in[6];
    const float* bg     = (const float*)d_in[7];
    const float* Wq_e   = (const float*)d_in[8];
    const float* Wk_e   = (const float*)d_in[9];
    const float* Wv_e   = (const float*)d_in[10];
    const float* Wo_e   = (const float*)d_in[11];
    const float* g_np   = (const float*)d_in[12];
    const float* b_np   = (const float*)d_in[13];
    const float* Wi_p   = (const float*)d_in[14];
    const float* bi_p   = (const float*)d_in[15];
    const float* Wo_p   = (const float*)d_in[16];
    const float* bo_p   = (const float*)d_in[17];
    const float* W_pe   = (const float*)d_in[18];
    const float* b_pe   = (const float*)d_in[19];
    const float* g_nh   = (const float*)d_in[20];
    const float* b_nh   = (const float*)d_in[21];
    const float* Wi_f   = (const float*)d_in[22];
    const float* bi_f   = (const float*)d_in[23];
    const float* Wo_f   = (const float*)d_in[24];
    const float* bo_f   = (const float*)d_in[25];
    const float* g_nfh  = (const float*)d_in[26];
    const float* b_nfh  = (const float*)d_in[27];
    const float* Wq_a   = (const float*)d_in[28];
    const float* bq_a   = (const float*)d_in[29];
    const float* Wk_a   = (const float*)d_in[30];
    const float* bk_a   = (const float*)d_in[31];
    const float* Wv_a   = (const float*)d_in[32];
    const float* bv_a   = (const float*)d_in[33];
    const float* Wd     = (const float*)d_in[34];
    const float* bd     = (const float*)d_in[35];
    const float* g_ln   = (const float*)d_in[36];
    const float* b_ln   = (const float*)d_in[37];
    float* out = (float*)d_out;

    const int Bc = 128, Sc = 128, Dc = 512, Fc = 84, Rc = 32, DSc = 64,
              NEc = 3, Hc = 8;
    const long long BS = (long long)Bc * Sc;  // 16384

    // ---- workspace layout (floats) ----
    float* w = (float*)d_ws;
    size_t off = 0;
    auto alloc = [&](size_t n) { float* p = w + off; off += n; return p; };
    float* xs    = alloc(BS * DSc);
    float* gate  = alloc(BS * NEc);
    float* qe    = alloc((size_t)NEc * BS * DSc);
    float* ke    = alloc((size_t)NEc * Bc * Rc * DSc);
    float* ve    = alloc((size_t)NEc * Bc * Rc * DSc);
    float* se    = alloc((size_t)NEc * BS * Rc);
    float* eo    = alloc((size_t)NEc * BS * DSc);
    float* eoW   = alloc((size_t)NEc * BS * DSc);
    float* pri   = alloc(BS * DSc);
    float* npri  = alloc(BS * DSc);
    float* pqkv  = alloc(BS * 3 * DSc);
    float* pattn = alloc(BS * DSc);
    float* pri2  = alloc(BS * DSc);
    float* priD  = alloc(BS * Dc);
    float* npriD = alloc(BS * Dc);
    float* fq    = alloc(BS * Dc);
    float* fk    = alloc(BS * Dc);
    float* fv    = alloc(BS * Dc);
    float* score = alloc((size_t)Bc * Hc * Sc * Sc);  // shared by all attn stages
    float* h2    = alloc(BS * Dc);
    float* nh    = alloc(BS * Dc);
    // safe reuses (lifetimes disjoint):
    float* aq    = fk;
    float* ak    = fv;
    float* av    = priD;
    float* fattn = fq;
    float* ctx   = npriD;
    float* y     = h2;

    auto gemm = [&](const float* A, const float* Wp, const float* bias,
                    const float* add, float* C, int M, int N, int K,
                    int lda, int ldw, int ldc, int transB, float scale,
                    int n1, int n2,
                    long long sA1, long long sA2, long long sW1, long long sW2,
                    long long sC1, long long sC2) {
        dim3 g((N + TN - 1) / TN, (M + TM - 1) / TM, n1 * n2);
        hipLaunchKernelGGL(wmma_gemm, g, dim3(512), 0, stream,
                           A, Wp, bias, add, C, M, N, K, lda, ldw, ldc,
                           transB, scale, n2, sA1, sA2, sW1, sW2, sC1, sC2);
    };

    const float sc64 = 0.125f;               // 1/sqrt(64)
    const float sc8  = 0.35355339059327373f; // 1/sqrt(8)

    // ---- Stage 1: MOE priori ----
    gemm(tag, W_moe, b_moe, nullptr, xs, (int)BS, DSc, Dc, Dc, DSc, DSc, 0, 1.f,
         1, 1, 0, 0, 0, 0, 0, 0);
    gemm(xs, Wg, bg, nullptr, gate, (int)BS, NEc, DSc, DSc, NEc, NEc, 0, 1.f,
         1, 1, 0, 0, 0, 0, 0, 0);
    softmax_rows<<<(int)BS, 128, 0, stream>>>(gate, NEc, 1, 0);

    gemm(xs, Wq_e, nullptr, nullptr, qe, (int)BS, DSc, DSc, DSc, DSc, DSc, 0, 1.f,
         NEc, 1, 0, 0, (long long)DSc * DSc, 0, BS * DSc, 0);
    gemm(refs, Wk_e, nullptr, nullptr, ke, Bc * Rc, DSc, Dc, Dc, DSc, DSc, 0, 1.f,
         NEc, 1, (long long)Bc * Rc * Dc, 0, (long long)Dc * DSc, 0,
         (long long)Bc * Rc * DSc, 0);
    gemm(refs, Wv_e, nullptr, nullptr, ve, Bc * Rc, DSc, Dc, Dc, DSc, DSc, 0, 1.f,
         NEc, 1, (long long)Bc * Rc * Dc, 0, (long long)Dc * DSc, 0,
         (long long)Bc * Rc * DSc, 0);
    // scores[e,b] = q[S,DS] @ k^T  -> [S,R]
    gemm(qe, ke, nullptr, nullptr, se, Sc, Rc, DSc, DSc, DSc, Rc, 1, sc64,
         NEc, Bc, BS * DSc, (long long)Sc * DSc,
         (long long)Bc * Rc * DSc, (long long)Rc * DSc,
         BS * Rc, (long long)Sc * Rc);
    softmax_rows<<<NEc * Bc * Sc, 128, 0, stream>>>(se, Rc, 1, 0);
    gemm(se, ve, nullptr, nullptr, eo, Sc, DSc, Rc, Rc, DSc, DSc, 0, 1.f,
         NEc, Bc, BS * Rc, (long long)Sc * Rc,
         (long long)Bc * Rc * DSc, (long long)Rc * DSc,
         BS * DSc, (long long)Sc * DSc);
    gemm(eo, Wo_e, nullptr, nullptr, eoW, (int)BS, DSc, DSc, DSc, DSc, DSc, 0, 1.f,
         NEc, 1, BS * DSc, 0, (long long)DSc * DSc, 0, BS * DSc, 0);
    moe_combine<<<(int)((BS * DSc + 255) / 256), 256, 0, stream>>>(
        eoW, gate, pri, (int)(BS * DSc), BS * DSc);

    // ---- Stage 2: priori causal self-attention (H=8, dh=8) ----
    ln_rows<<<(int)BS, 256, 0, stream>>>(pri, g_np, b_np, npri, DSc, 1e-5f);
    gemm(npri, Wi_p, bi_p, nullptr, pqkv, (int)BS, 3 * DSc, DSc, DSc, 3 * DSc,
         3 * DSc, 0, 1.f, 1, 1, 0, 0, 0, 0, 0, 0);
    gemm(pqkv, pqkv + DSc, nullptr, nullptr, score, Sc, Sc, 8, 3 * DSc, 3 * DSc,
         Sc, 1, sc8, Bc, Hc, (long long)Sc * 3 * DSc, 8,
         (long long)Sc * 3 * DSc, 8, (long long)Hc * Sc * Sc, (long long)Sc * Sc);
    softmax_rows<<<Bc * Hc * Sc, 128, 0, stream>>>(score, Sc, Sc, 1);
    gemm(score, pqkv + 2 * DSc, nullptr, nullptr, pattn, Sc, 8, Sc, Sc, 3 * DSc,
         DSc, 0, 1.f, Bc, Hc, (long long)Hc * Sc * Sc, (long long)Sc * Sc,
         (long long)Sc * 3 * DSc, 8, (long long)Sc * DSc, 8);
    gemm(pattn, Wo_p, bo_p, pri, pri2, (int)BS, DSc, DSc, DSc, DSc, DSc, 0, 1.f,
         1, 1, 0, 0, 0, 0, 0, 0);
    gemm(pri2, W_pe, b_pe, nullptr, priD, (int)BS, Dc, DSc, DSc, Dc, Dc, 0, 1.f,
         1, 1, 0, 0, 0, 0, 0, 0);

    // ---- Stage 3: fusion cross-attention (H=8, dh=64, causal) ----
    ln_rows<<<(int)BS, 256, 0, stream>>>(priD, g_nh, b_nh, npriD, Dc, 1e-5f);
    gemm(hidden, Wi_f, bi_f, nullptr, fq, (int)BS, Dc, Dc, Dc, 3 * Dc, Dc, 0, 1.f,
         1, 1, 0, 0, 0, 0, 0, 0);
    gemm(npriD, Wi_f + Dc, bi_f + Dc, nullptr, fk, (int)BS, Dc, Dc, Dc, 3 * Dc,
         Dc, 0, 1.f, 1, 1, 0, 0, 0, 0, 0, 0);
    gemm(npriD, Wi_f + 2 * Dc, bi_f + 2 * Dc, nullptr, fv, (int)BS, Dc, Dc, Dc,
         3 * Dc, Dc, 0, 1.f, 1, 1, 0, 0, 0, 0, 0, 0);
    gemm(fq, fk, nullptr, nullptr, score, Sc, Sc, 64, Dc, Dc, Sc, 1, sc64,
         Bc, Hc, (long long)Sc * Dc, 64, (long long)Sc * Dc, 64,
         (long long)Hc * Sc * Sc, (long long)Sc * Sc);
    softmax_rows<<<Bc * Hc * Sc, 128, 0, stream>>>(score, Sc, Sc, 1);
    gemm(score, fv, nullptr, nullptr, fattn, Sc, 64, Sc, Sc, Dc, Dc, 0, 1.f,
         Bc, Hc, (long long)Hc * Sc * Sc, (long long)Sc * Sc,
         (long long)Sc * Dc, 64, (long long)Sc * Dc, 64);
    gemm(fattn, Wo_f, bo_f, hidden, h2, (int)BS, Dc, Dc, Dc, Dc, Dc, 0, 1.f,
         1, 1, 0, 0, 0, 0, 0, 0);

    // ---- Stage 4: feat cross-attention + dense + post-LN ----
    ln_rows<<<(int)BS, 256, 0, stream>>>(h2, g_nfh, b_nfh, nh, Dc, 1e-5f);
    gemm(nh, Wq_a, bq_a, nullptr, aq, (int)BS, Dc, Dc, Dc, Dc, Dc, 0, 1.f,
         1, 1, 0, 0, 0, 0, 0, 0);
    gemm(feats, Wk_a, bk_a, nullptr, ak, Bc * Fc, Dc, Dc, Dc, Dc, Dc, 0, 1.f,
         1, 1, 0, 0, 0, 0, 0, 0);
    gemm(feats, Wv_a, bv_a, nullptr, av, Bc * Fc, Dc, Dc, Dc, Dc, Dc, 0, 1.f,
         1, 1, 0, 0, 0, 0, 0, 0);
    gemm(aq, ak, nullptr, nullptr, score, Sc, Fc, 64, Dc, Dc, Fc, 1, sc64,
         Bc, Hc, (long long)Sc * Dc, 64, (long long)Fc * Dc, 64,
         (long long)Hc * Sc * Fc, (long long)Sc * Fc);
    softmax_rows<<<Bc * Hc * Sc, 128, 0, stream>>>(score, Fc, 1, 0);
    gemm(score, av, nullptr, nullptr, ctx, Sc, 64, Fc, Fc, Dc, Dc, 0, 1.f,
         Bc, Hc, (long long)Hc * Sc * Fc, (long long)Sc * Fc,
         (long long)Fc * Dc, 64, (long long)Sc * Dc, 64);
    gemm(ctx, Wd, bd, nh, y, (int)BS, Dc, Dc, Dc, Dc, Dc, 0, 1.f,
         1, 1, 0, 0, 0, 0, 0, 0);
    ln_rows<<<(int)BS, 256, 0, stream>>>(y, g_ln, b_ln, out, Dc, 1e-12f);
}
